// EGSleepNet_8315056685660
// MI455X (gfx1250) — matrix operand order repaired and checked by reading, not compiled, not run on hardware
//
#include <hip/hip_runtime.h>
#include <hip/hip_bf16.h>

typedef __attribute__((ext_vector_type(16))) __bf16 v16bf;
typedef __attribute__((ext_vector_type(8)))  __bf16 v8bf;
typedef __attribute__((ext_vector_type(8)))  float  v8f;
typedef __attribute__((ext_vector_type(4)))  float  f4;

#define BB   2048
#define SS   29
#define DD   128
#define NHH  8
#define FFF  512
#define D3   384
#define NN   30
#define EE   5
#define GHH  4

// ---------------------------------------------------------------------------
// Tiled WMMA GEMM:  C[M,N] = act(A[M,K] @ W[K,N] [+ bias])
// Requirements (guaranteed by all launches): M%128==0, N%128==0, K%32==0.
// fp32 in HBM, bf16 staged in LDS, fp32 accumulate, v_wmma_f32_16x16x32_bf16.
// Block tile 128x128 (256 threads / 8 waves); each wave owns a 64x32 patch
// -> 8 WMMA per K-step from 4 A-frags + 2 B-frags (12 ds_load_b128).
// LDS rows padded to 40 elems (80B, mult of 16) so frag reads are b128.
// ---------------------------------------------------------------------------
template <int ACT, int HAS_BIAS>
__global__ __launch_bounds__(256)
void gemm_bf16_wmma(const float* __restrict__ A, const float* __restrict__ W,
                    const float* __restrict__ bias, float* __restrict__ C,
                    int M, int N, int K)
{
    __shared__ __bf16 As[128][40];   // [m][k]
    __shared__ __bf16 Bs[128][40];   // [n][k]  (transposed tile)

    const int tid  = threadIdx.x;
    const int bm   = blockIdx.x * 128;
    const int bn   = blockIdx.y * 128;
    const int wave = tid >> 5;
    const int lane = tid & 31;
    const int m0   = (wave & 1) * 64;    // wave M patch (64 rows)
    const int n0   = (wave >> 1) * 32;   // wave N patch (32 cols)

    // A staging: 128 rows x 32 k; thread -> row tid>>1, 16 consecutive k
    const int la_m = tid >> 1;
    const int la_c = (tid & 1) * 16;
    // B staging: 128 n x 32 k; thread -> n = tid&127, 16 consecutive k
    const int lb_n = tid & 127;
    const int lb_k = (tid >> 7) * 16;

    // fragment addressing
    const int fm  = lane & 15;
    const int fka = (lane >> 4) * 8;    // A: k-half offset (elems 0..7 <-> k=fka+e)
    const int fkb = (lane >> 4) * 8;    // B: k-half offset (same as A)

    v8f acc[4][2];
#pragma unroll
    for (int i = 0; i < 4; ++i)
#pragma unroll
        for (int j = 0; j < 2; ++j) acc[i][j] = (v8f){};

    for (int kk = 0; kk < K; kk += 32) {
        // ---- stage A (4x global_load_b128, 2x ds_store_b128) ----
        const float* ap = &A[(size_t)(bm + la_m) * K + kk + la_c];
        f4 a0 = *(const f4*)ap;
        f4 a1 = *(const f4*)(ap + 4);
        f4 a2 = *(const f4*)(ap + 8);
        f4 a3 = *(const f4*)(ap + 12);
        v8bf av0, av1;
#pragma unroll
        for (int j = 0; j < 4; ++j) {
            av0[j] = (__bf16)a0[j]; av0[4 + j] = (__bf16)a1[j];
            av1[j] = (__bf16)a2[j]; av1[4 + j] = (__bf16)a3[j];
        }
        *(v8bf*)&As[la_m][la_c]     = av0;
        *(v8bf*)&As[la_m][la_c + 8] = av1;

        // ---- stage B transposed (coalesced b32 column reads, 2x ds_store_b128) ----
        v8bf bv0, bv1;
#pragma unroll
        for (int j = 0; j < 8; ++j) {
            bv0[j] = (__bf16)W[(size_t)(kk + lb_k + j) * N + bn + lb_n];
            bv1[j] = (__bf16)W[(size_t)(kk + lb_k + 8 + j) * N + bn + lb_n];
        }
        *(v8bf*)&Bs[lb_n][lb_k]     = bv0;
        *(v8bf*)&Bs[lb_n][lb_k + 8] = bv1;
        __syncthreads();

        // ---- fragments: all 16B-aligned ds_load_b128 ----
        v16bf af[4], bf[2];
#pragma unroll
        for (int i = 0; i < 4; ++i) {
            v8bf x0 = *(const v8bf*)&As[m0 + i * 16 + fm][fka];
            v8bf x1 = *(const v8bf*)&As[m0 + i * 16 + fm][16 + fka];
            af[i] = __builtin_shufflevector(x0, x1, 0,1,2,3,4,5,6,7,8,9,10,11,12,13,14,15);
        }
#pragma unroll
        for (int j = 0; j < 2; ++j) {
            v8bf y0 = *(const v8bf*)&Bs[n0 + j * 16 + fm][fkb];
            v8bf y1 = *(const v8bf*)&Bs[n0 + j * 16 + fm][16 + fkb];
            bf[j] = __builtin_shufflevector(y0, y1, 0,1,2,3,4,5,6,7,8,9,10,11,12,13,14,15);
        }

#pragma unroll
        for (int i = 0; i < 4; ++i)
#pragma unroll
            for (int j = 0; j < 2; ++j)
                acc[i][j] = __builtin_amdgcn_wmma_f32_16x16x32_bf16(
                    false, af[i], false, bf[j], (short)0, acc[i][j], false, false);
        __syncthreads();
    }

    // ---- epilogue: D layout = lane col (lane&15), rows (lane>>4)*8 + r ----
    const int mh = (lane >> 4) * 8;
    const int cn = bn + n0 + (lane & 15);
#pragma unroll
    for (int j = 0; j < 2; ++j) {
        const int col = cn + j * 16;
        float bv = 0.f;
        if (HAS_BIAS) bv = bias[col];
#pragma unroll
        for (int i = 0; i < 4; ++i) {
#pragma unroll
            for (int r = 0; r < 8; ++r) {
                int row = bm + m0 + i * 16 + mh + r;
                float v = acc[i][j][r] + bv;
                if (ACT == 1) v = v > 0.f ? v : 0.f;
                C[(size_t)row * N + col] = v;
            }
        }
    }
}

// ---------------------------------------------------------------------------
// Elementwise / small kernels
// ---------------------------------------------------------------------------
__global__ void split_pe_k(const float* __restrict__ x, const float* __restrict__ pe,
                           float* x1, float* x2, float* x3)
{
    size_t idx = (size_t)blockIdx.x * blockDim.x + threadIdx.x;
    size_t total = (size_t)BB * 3 * SS * DD;
    if (idx >= total) return;
    int d = idx % DD;
    int s = (idx / DD) % SS;
    int m = (idx / ((size_t)DD * SS)) % 3;
    size_t b = idx / ((size_t)DD * SS * 3);
    float v = x[idx] + pe[s * DD + d];
    float* dst = (m == 0) ? x1 : (m == 1) ? x2 : x3;
    dst[(b * SS + s) * DD + d] = v;
}

__global__ void concat2_k(const float* __restrict__ a, const float* __restrict__ b,
                          float* __restrict__ cat, int rows)
{
    size_t idx = (size_t)blockIdx.x * blockDim.x + threadIdx.x;
    size_t total = (size_t)rows * 256;
    if (idx >= total) return;
    int c = idx % 256;
    size_t row = idx / 256;
    cat[idx] = (c < 128) ? a[row * 128 + c] : b[row * 128 + (c - 128)];
}

__global__ void concat3_k(const float* __restrict__ a, const float* __restrict__ b,
                          const float* __restrict__ c3, float* __restrict__ cat, int rows)
{
    size_t idx = (size_t)blockIdx.x * blockDim.x + threadIdx.x;
    size_t total = (size_t)rows * D3;
    if (idx >= total) return;
    int c = idx % D3;
    size_t row = idx / D3;
    float v;
    if (c < 128)      v = a[row * 128 + c];
    else if (c < 256) v = b[row * 128 + (c - 128)];
    else              v = c3[row * 128 + (c - 256)];
    cat[idx] = v;
}

// mode 0: x = LN(x + h)        (h may be null -> LN(x))
// mode 1: x += LN(relu(h))
__global__ __launch_bounds__(128)
void ln_kernel(float* __restrict__ x, const float* __restrict__ h,
               const float* __restrict__ g, const float* __restrict__ bb,
               int D, int mode)
{
    const int row = blockIdx.x;
    const int tid = threadIdx.x;
    __shared__ float vals[384];
    __shared__ float red[128];
    const size_t base = (size_t)row * D;

    float lsum = 0.f;
    for (int d = tid; d < D; d += 128) {
        float v;
        if (mode == 0) v = x[base + d] + (h ? h[base + d] : 0.f);
        else           { v = h[base + d]; v = v > 0.f ? v : 0.f; }
        vals[d] = v; lsum += v;
    }
    red[tid] = lsum; __syncthreads();
    for (int s = 64; s > 0; s >>= 1) { if (tid < s) red[tid] += red[tid + s]; __syncthreads(); }
    const float mean = red[0] / D; __syncthreads();

    float lvar = 0.f;
    for (int d = tid; d < D; d += 128) { float t = vals[d] - mean; lvar += t * t; }
    red[tid] = lvar; __syncthreads();
    for (int s = 64; s > 0; s >>= 1) { if (tid < s) red[tid] += red[tid + s]; __syncthreads(); }
    const float inv = rsqrtf(red[0] / D + 1e-5f);

    for (int d = tid; d < D; d += 128) {
        float r = (vals[d] - mean) * inv * g[d] + bb[d];
        if (mode == 0) x[base + d] = r; else x[base + d] += r;
    }
}

__global__ __launch_bounds__(32)
void attn_kernel(const float* __restrict__ q, const float* __restrict__ k,
                 const float* __restrict__ v, float* __restrict__ o)
{
    const int bh = blockIdx.x;
    const int b = bh / NHH, hh = bh % NHH;
    const int col0 = hh * 16;
    __shared__ float qs[SS][16], ks[SS][16], vs[SS][16];
    const int lane = threadIdx.x;
    for (int idx = lane; idx < SS * 16; idx += 32) {
        int s = idx >> 4, d = idx & 15;
        size_t g = ((size_t)(b * SS + s)) * DD + col0 + d;
        qs[s][d] = q[g]; ks[s][d] = k[g]; vs[s][d] = v[g];
    }
    __syncthreads();
    if (lane < SS) {
        float lg[SS]; float mx = -1e30f;
        for (int j = 0; j < SS; ++j) {
            float dd = 0.f;
            for (int t = 0; t < 16; ++t) dd += qs[lane][t] * ks[j][t];
            dd *= 0.25f; lg[j] = dd; mx = fmaxf(mx, dd);
        }
        float sum = 0.f;
        for (int j = 0; j < SS; ++j) { float e = __expf(lg[j] - mx); lg[j] = e; sum += e; }
        const float inv = 1.f / sum;
        float ov[16];
        for (int t = 0; t < 16; ++t) ov[t] = 0.f;
        for (int j = 0; j < SS; ++j) {
            float a = lg[j] * inv;
            for (int t = 0; t < 16; ++t) ov[t] += a * vs[j][t];
        }
        size_t g = ((size_t)(b * SS + lane)) * DD + col0;
        for (int t = 0; t < 16; ++t) o[g + t] = ov[t];
    }
}

// conv1d over [B, C=29, L=384], kernel 3, pad 1; y = (conv + bias)*gamma + beta, opt relu
__global__ void conv29_k(const float* __restrict__ x, const float* __restrict__ w,
                         const float* __restrict__ bias, const float* __restrict__ gamma,
                         const float* __restrict__ beta, float* __restrict__ y, int do_relu)
{
    size_t idx = (size_t)blockIdx.x * blockDim.x + threadIdx.x;
    size_t total = (size_t)BB * SS * D3;
    if (idx >= total) return;
    int l = idx % D3;
    int o = (idx / D3) % SS;
    size_t b = idx / ((size_t)D3 * SS);
    float acc = bias[o];
    for (int i = 0; i < SS; ++i) {
        const float* xr = x + (b * SS + i) * D3;
        const float* wr = w + (o * SS + i) * 3;
        float lft = (l > 0)       ? xr[l - 1] : 0.f;
        float mid = xr[l];
        float rgt = (l < D3 - 1)  ? xr[l + 1] : 0.f;
        acc += lft * wr[0] + mid * wr[1] + rgt * wr[2];
    }
    acc = acc * gamma[o] + beta[o];
    if (do_relu) acc = acc > 0.f ? acc : 0.f;
    y[idx] = acc;
}

__global__ __launch_bounds__(128)
void row_mean_k(const float* __restrict__ x, float* __restrict__ out, int D)
{
    const int row = blockIdx.x, tid = threadIdx.x;
    __shared__ float red[128];
    float ls = 0.f;
    for (int d = tid; d < D; d += 128) ls += x[(size_t)row * D + d];
    red[tid] = ls; __syncthreads();
    for (int s = 64; s > 0; s >>= 1) { if (tid < s) red[tid] += red[tid + s]; __syncthreads(); }
    if (tid == 0) out[row] = red[0] / D;
}

__global__ __launch_bounds__(32)
void fca_attn_k(const float* __restrict__ s, const float* __restrict__ fa1,
                const float* __restrict__ fa2, float* __restrict__ w)
{
    const int b = blockIdx.x;
    __shared__ float hid;
    if (threadIdx.x == 0) {
        float acc = 0.f;
        for (int c = 0; c < SS; ++c) acc += s[b * SS + c] * fa1[c];
        hid = fmaxf(acc, 0.f);
    }
    __syncthreads();
    if (threadIdx.x < SS) {
        float v = hid * fa2[threadIdx.x];
        w[b * SS + threadIdx.x] = 1.f / (1.f + __expf(-v));
    }
}

__global__ void fca_out_k(const float* __restrict__ h2, const float* __restrict__ wch,
                          float* __restrict__ x)
{
    size_t idx = (size_t)blockIdx.x * blockDim.x + threadIdx.x;
    size_t total = (size_t)BB * SS * D3;
    if (idx >= total) return;
    size_t bc = idx / D3;                 // b*29 + c
    float v = h2[idx] * wch[bc] + x[idx];
    x[idx] = v > 0.f ? v : 0.f;
}

__global__ void append_mean_k(const float* __restrict__ h, float* __restrict__ g0)
{
    size_t idx = (size_t)blockIdx.x * blockDim.x + threadIdx.x;
    size_t total = (size_t)BB * D3;
    if (idx >= total) return;
    int d = idx % D3;
    size_t b = idx / D3;
    float sum = 0.f;
    for (int s = 0; s < SS; ++s) {
        float v = h[(b * SS + s) * D3 + d];
        g0[(b * NN + s) * D3 + d] = v;
        sum += v;
    }
    g0[(b * NN + SS) * D3 + d] = sum / SS;
}

__global__ __launch_bounds__(128)
void proto_norm_k(const float* __restrict__ proto, float* __restrict__ pn)
{
    const int e = blockIdx.x, tid = threadIdx.x;
    __shared__ float red[128];
    float ls = 0.f;
    for (int d = tid; d < D3; d += 128) { float v = proto[e * D3 + d]; ls += v * v; }
    red[tid] = ls; __syncthreads();
    for (int s = 64; s > 0; s >>= 1) { if (tid < s) red[tid] += red[tid + s]; __syncthreads(); }
    float inv = 1.f / (sqrtf(red[0]) + 1e-8f);
    for (int d = tid; d < D3; d += 128) pn[e * D3 + d] = proto[e * D3 + d] * inv;
}

__global__ __launch_bounds__(128)
void hyper_H_k(const float* __restrict__ x, const float* __restrict__ pn,
               float* __restrict__ H)
{
    const int row = blockIdx.x;   // b*30 + n
    const int tid = threadIdx.x;
    __shared__ float red[128];
    const size_t base = (size_t)row * D3;
    float ls = 0.f;
    for (int d = tid; d < D3; d += 128) { float v = x[base + d]; ls += v * v; }
    red[tid] = ls; __syncthreads();
    for (int s = 64; s > 0; s >>= 1) { if (tid < s) red[tid] += red[tid + s]; __syncthreads(); }
    const float nx = sqrtf(red[0]) + 1e-8f; __syncthreads();
    for (int e = 0; e < EE; ++e) {
        float ld = 0.f;
        for (int d = tid; d < D3; d += 128) ld += x[base + d] * pn[e * D3 + d];
        red[tid] = ld; __syncthreads();
        for (int s = 64; s > 0; s >>= 1) { if (tid < s) red[tid] += red[tid + s]; __syncthreads(); }
        if (tid == 0) H[(size_t)row * EE + e] = fmaxf(red[0] / nx, 0.f);
        __syncthreads();
    }
}

__global__ void deg_e_k(const float* __restrict__ H, float* __restrict__ De)
{
    size_t idx = (size_t)blockIdx.x * blockDim.x + threadIdx.x;
    if (idx >= (size_t)BB * EE) return;
    int e = idx % EE; size_t b = idx / EE;
    float s = 1e-6f;
    for (int n = 0; n < NN; ++n) s += H[(b * NN + n) * EE + e];
    De[idx] = s;
}

__global__ void deg_v_k(const float* __restrict__ H, float* __restrict__ Dv)
{
    size_t idx = (size_t)blockIdx.x * blockDim.x + threadIdx.x;
    if (idx >= (size_t)BB * NN) return;
    float s = 1e-6f;
    for (int e = 0; e < EE; ++e) s += H[idx * EE + e];
    Dv[idx] = s;
}

__global__ void edge_k(const float* __restrict__ H, const float* __restrict__ De,
                       const float* __restrict__ x, float* __restrict__ edge)
{
    size_t idx = (size_t)blockIdx.x * blockDim.x + threadIdx.x;
    size_t total = (size_t)BB * EE * D3;
    if (idx >= total) return;
    int d = idx % D3;
    int e = (idx / D3) % EE;
    size_t b = idx / ((size_t)D3 * EE);
    float acc = 0.f;
    for (int n = 0; n < NN; ++n)
        acc += H[(b * NN + n) * EE + e] * x[(b * NN + n) * (size_t)D3 + d];
    edge[idx] = acc / De[b * EE + e];
}

__global__ void hgout_k(const float* __restrict__ H, const float* __restrict__ edge,
                        const float* __restrict__ Dv, float* __restrict__ out)
{
    size_t idx = (size_t)blockIdx.x * blockDim.x + threadIdx.x;
    size_t total = (size_t)BB * NN * D3;
    if (idx >= total) return;
    int d = idx % D3;
    int n = (idx / D3) % NN;
    size_t b = idx / ((size_t)D3 * NN);
    float acc = 0.f;
    for (int e = 0; e < EE; ++e)
        acc += H[(b * NN + n) * EE + e] * edge[(b * EE + e) * (size_t)D3 + d];
    out[idx] = acc / Dv[b * NN + n];
}

__global__ void adj_k(const float* __restrict__ H, float* __restrict__ Adj)
{
    size_t idx = (size_t)blockIdx.x * blockDim.x + threadIdx.x;
    if (idx >= (size_t)BB * NN * NN) return;
    int j = idx % NN;
    int i = (idx / NN) % NN;
    size_t b = idx / (NN * NN);
    float a = (i == j) ? 1.f : 0.f;
    if (a == 0.f) {
        for (int e = 0; e < EE; ++e)
            if (H[(b * NN + i) * EE + e] > 0.f && H[(b * NN + j) * EE + e] > 0.f) { a = 1.f; break; }
    }
    Adj[idx] = a;
}

template <int HEADS, int DH>
__global__ __launch_bounds__(128)
void gat_attn_k(const float* __restrict__ h, const float* __restrict__ Adj,
                const float* __restrict__ a_src, const float* __restrict__ a_dst,
                const float* __restrict__ bias, float* __restrict__ out)
{
    const int b = blockIdx.x / HEADS, hh = blockIdx.x % HEADS;
    __shared__ float hs[NN][DH];
    __shared__ float es[NN], ed[NN];
    __shared__ float alph[NN][NN];
    const int tid = threadIdx.x;

    for (int idx = tid; idx < NN * DH; idx += 128) {
        int n = idx / DH, d = idx % DH;
        hs[n][d] = h[((size_t)(b * NN + n)) * D3 + hh * DH + d];
    }
    __syncthreads();
    for (int n = tid; n < NN; n += 128) {
        float s1 = 0.f, s2 = 0.f;
        for (int d = 0; d < DH; ++d) { s1 += hs[n][d] * a_src[hh * DH + d]; s2 += hs[n][d] * a_dst[hh * DH + d]; }
        es[n] = s1; ed[n] = s2;
    }
    __syncthreads();
    for (int idx = tid; idx < NN * NN; idx += 128) {
        int i = idx / NN, j = idx % NN;
        float l = ed[i] + es[j];
        l = l > 0.f ? l : 0.2f * l;
        if (Adj[(size_t)b * NN * NN + i * NN + j] == 0.f) l = -1e9f;
        alph[i][j] = l;
    }
    __syncthreads();
    for (int i = tid; i < NN; i += 128) {
        float mx = -1e30f;
        for (int j = 0; j < NN; ++j) mx = fmaxf(mx, alph[i][j]);
        float sum = 0.f;
        for (int j = 0; j < NN; ++j) { float e = __expf(alph[i][j] - mx); alph[i][j] = e; sum += e; }
        float inv = 1.f / sum;
        for (int j = 0; j < NN; ++j) alph[i][j] *= inv;
    }
    __syncthreads();
    for (int idx = tid; idx < NN * DH; idx += 128) {
        int i = idx / DH, d = idx % DH;
        float acc = 0.f;
        for (int j = 0; j < NN; ++j) acc += alph[i][j] * hs[j][d];
        acc += bias[hh * DH + d];
        acc = acc > 0.f ? acc : (__expf(acc) - 1.f);   // ELU
        out[((size_t)(b * NN + i)) * D3 + hh * DH + d] = acc;
    }
}

__global__ void combine_k(const float* __restrict__ g0, const float* __restrict__ r1,
                          float* __restrict__ o)
{
    size_t idx = (size_t)blockIdx.x * blockDim.x + threadIdx.x;
    if (idx >= (size_t)BB * NN * D3) return;
    o[idx] += g0[idx] + r1[idx];
}

__global__ void pool_mean_k(const float* __restrict__ x, float* __restrict__ out)
{
    size_t idx = (size_t)blockIdx.x * blockDim.x + threadIdx.x;
    if (idx >= (size_t)BB * D3) return;
    int d = idx % D3;
    size_t b = idx / D3;
    float s = 0.f;
    for (int n = 0; n < NN; ++n) s += x[(b * NN + n) * (size_t)D3 + d];
    out[idx] = s / NN;
}

__global__ void fc2_k(const float* __restrict__ f, const float* __restrict__ w,
                      const float* __restrict__ b, float* __restrict__ out)
{
    size_t idx = (size_t)blockIdx.x * blockDim.x + threadIdx.x;
    if (idx >= (size_t)BB * 5) return;
    int c = idx % 5;
    size_t bb = idx / 5;
    float acc = b[c];
    for (int k = 0; k < 128; ++k) acc += f[bb * 128 + k] * w[k * 5 + c];
    out[idx] = acc;
}

// ---------------------------------------------------------------------------
extern "C" void kernel_launch(void* const* d_in, const int* in_sizes, int n_in,
                              void* d_out, int out_size, void* d_ws, size_t ws_size,
                              hipStream_t stream)
{
    (void)in_sizes; (void)n_in; (void)out_size; (void)ws_size;
    auto P = [&](int i) { return (const float*)d_in[i]; };

    const float* x_in = P(0);
    const float* pe   = P(1);
    const float* fw = P(2); const float* fb = P(3); const float* fg = P(4); const float* fbe = P(5);
    const int dctB = 6 + 3 * 32;          // 102
    const float* cw1 = P(dctB + 0); const float* cb1 = P(dctB + 1);
    const float* cg1 = P(dctB + 2); const float* cbe1 = P(dctB + 3);
    const float* cw2 = P(dctB + 4); const float* cb2 = P(dctB + 5);
    const float* cg2 = P(dctB + 6); const float* cbe2 = P(dctB + 7);
    const float* fa1 = P(dctB + 8); const float* fa2 = P(dctB + 9);
    const float* lng = P(112); const float* lnb = P(113);
    const float* proto = P(114); const float* theta = P(115); const float* thb = P(116);
    const float* g1w = P(117); const float* g1as = P(118); const float* g1ad = P(119); const float* g1b = P(120);
    const float* g2w = P(121); const float* g2as = P(122); const float* g2ad = P(123); const float* g2b = P(124);
    const float* fc1w = P(125); const float* fc1b = P(126);
    const float* fc2w = P(127); const float* fc2b = P(128);

    // ------ workspace arena (floats) ------
    const size_t BSD  = (size_t)BB * SS * DD;
    const size_t BS2D = 2 * BSD;
    const size_t BSFF = (size_t)BB * SS * FFF;
    const size_t BS3D = (size_t)BB * SS * D3;
    const size_t BN3D = (size_t)BB * NN * D3;
    float* Wp = (float*)d_ws;
    size_t off = 0;
    auto alloc = [&](size_t n) { float* p = Wp + off; off += n; return p; };
    float* x1   = alloc(BSD);
    float* x2   = alloc(BSD);
    float* x3   = alloc(BSD);
    float* cat  = alloc(BS2D);
    float* ftmp = alloc(BSD);
    float* qb   = alloc(BSD);
    float* kb   = alloc(BSD);
    float* vb   = alloc(BSD);
    float* ao   = alloc(BSD);
    float* ff   = alloc(BSFF);
    float* hcat = alloc(BS3D);
    float* h1   = alloc(BS3D);
    float* h2   = alloc(BS3D);
    float* sq   = alloc((size_t)BB * SS);
    float* wch  = alloc((size_t)BB * SS);
    float* g0   = alloc(BN3D);
    float* Hh   = alloc((size_t)BB * NN * EE);
    float* De   = alloc((size_t)BB * EE);
    float* Dv   = alloc((size_t)BB * NN);
    float* edg  = alloc((size_t)BB * EE * D3);
    float* hg   = alloc(BN3D);
    float* r1   = alloc(BN3D);
    float* Adj  = alloc((size_t)BB * NN * NN);
    float* g1h  = alloc(BN3D);
    float* g1o  = alloc(BN3D);
    float* g2h  = alloc(BN3D);
    float* g2o  = alloc(BN3D);
    float* pool = alloc((size_t)BB * D3);
    float* f1   = alloc((size_t)BB * 128);
    float* pn   = alloc((size_t)EE * D3);

    const int BS = BB * SS;     // 59392 rows (mult of 128)
    const int BN = BB * NN;     // 61440 rows (mult of 128)
    auto EW = [&](size_t n) { return dim3((unsigned)((n + 255) / 256)); };
    auto GG = [&](int M, int N) { return dim3((unsigned)(M / 128), (unsigned)(N / 128)); };

    // 1. split input + positional encoding
    split_pe_k<<<EW((size_t)BB * 3 * SS * DD), 256, 0, stream>>>(x_in, pe, x1, x2, x3);

    // 2. fusion (sequential: x1, then x2, then x3)
    auto fusion = [&](float* target, const float* a, const float* b) {
        concat2_k<<<EW((size_t)BS * 256), 256, 0, stream>>>(a, b, cat, BS);
        gemm_bf16_wmma<0, 1><<<GG(BS, 128), 256, 0, stream>>>(cat, fw, fb, ftmp, BS, 128, 256);
        ln_kernel<<<BS, 128, 0, stream>>>(target, ftmp, fg, fbe, 128, 1);
    };
    fusion(x1, x2, x3);
    fusion(x2, x1, x3);
    fusion(x3, x1, x2);

    // 3. transformer encoders
    float* xb[3] = { x1, x2, x3 };
    for (int e = 0; e < 3; ++e) {
        float* xe = xb[e];
        for (int l = 0; l < 2; ++l) {
            const int p0 = 6 + e * 32 + l * 16;
            gemm_bf16_wmma<0, 1><<<GG(BS, 128), 256, 0, stream>>>(xe, P(p0 + 0), P(p0 + 1), qb, BS, 128, 128);
            gemm_bf16_wmma<0, 1><<<GG(BS, 128), 256, 0, stream>>>(xe, P(p0 + 2), P(p0 + 3), kb, BS, 128, 128);
            gemm_bf16_wmma<0, 1><<<GG(BS, 128), 256, 0, stream>>>(xe, P(p0 + 4), P(p0 + 5), vb, BS, 128, 128);
            attn_kernel<<<BB * NHH, 32, 0, stream>>>(qb, kb, vb, ao);
            gemm_bf16_wmma<0, 1><<<GG(BS, 128), 256, 0, stream>>>(ao, P(p0 + 6), P(p0 + 7), ftmp, BS, 128, 128);
            ln_kernel<<<BS, 128, 0, stream>>>(xe, ftmp, P(p0 + 8), P(p0 + 9), 128, 0);
            gemm_bf16_wmma<1, 1><<<GG(BS, 512), 256, 0, stream>>>(xe, P(p0 + 10), P(p0 + 11), ff, BS, 512, 128);
            gemm_bf16_wmma<0, 1><<<GG(BS, 128), 256, 0, stream>>>(ff, P(p0 + 12), P(p0 + 13), ftmp, BS, 128, 512);
            ln_kernel<<<BS, 128, 0, stream>>>(xe, ftmp, P(p0 + 14), P(p0 + 15), 128, 0);
        }
    }

    // 4. concat -> FCA block
    concat3_k<<<EW((size_t)BS * D3), 256, 0, stream>>>(x1, x2, x3, hcat, BS);
    conv29_k<<<EW(BS3D), 256, 0, stream>>>(hcat, cw1, cb1, cg1, cbe1, h1, 1);
    conv29_k<<<EW(BS3D), 256, 0, stream>>>(h1, cw2, cb2, cg2, cbe2, h2, 0);
    row_mean_k<<<BS, 128, 0, stream>>>(h2, sq, D3);
    fca_attn_k<<<BB, 32, 0, stream>>>(sq, fa1, fa2, wch);
    fca_out_k<<<EW(BS3D), 256, 0, stream>>>(h2, wch, hcat);
    ln_kernel<<<BS, 128, 0, stream>>>(hcat, nullptr, lng, lnb, D3, 0);

    // 5. append mean node -> g0 (residual)
    append_mean_k<<<EW((size_t)BB * D3), 256, 0, stream>>>(hcat, g0);

    // 6. hypergraph
    proto_norm_k<<<EE, 128, 0, stream>>>(proto, pn);
    hyper_H_k<<<BN, 128, 0, stream>>>(g0, pn, Hh);
    deg_e_k<<<EW((size_t)BB * EE), 256, 0, stream>>>(Hh, De);
    deg_v_k<<<EW((size_t)BB * NN), 256, 0, stream>>>(Hh, Dv);
    edge_k<<<EW((size_t)BB * EE * D3), 256, 0, stream>>>(Hh, De, g0, edg);
    hgout_k<<<EW(BN3D), 256, 0, stream>>>(Hh, edg, Dv, hg);
    gemm_bf16_wmma<1, 1><<<GG(BN, D3), 256, 0, stream>>>(hg, theta, thb, r1, BN, D3, D3);
    adj_k<<<EW((size_t)BB * NN * NN), 256, 0, stream>>>(Hh, Adj);

    // 7. GAT layers
    gemm_bf16_wmma<0, 0><<<GG(BN, D3), 256, 0, stream>>>(r1, g1w, nullptr, g1h, BN, D3, D3);
    gat_attn_k<GHH, 96><<<BB * GHH, 128, 0, stream>>>(g1h, Adj, g1as, g1ad, g1b, g1o);
    gemm_bf16_wmma<0, 0><<<GG(BN, D3), 256, 0, stream>>>(g1o, g2w, nullptr, g2h, BN, D3, D3);
    gat_attn_k<1, D3><<<BB, 128, 0, stream>>>(g2h, Adj, g2as, g2ad, g2b, g2o);

    // 8. combine residuals, pool, head
    combine_k<<<EW(BN3D), 256, 0, stream>>>(g0, r1, g2o);
    pool_mean_k<<<EW((size_t)BB * D3), 256, 0, stream>>>(g2o, pool);
    gemm_bf16_wmma<1, 1><<<GG(BB, 128), 256, 0, stream>>>(pool, fc1w, fc1b, f1, BB, 128, D3);
    fc2_k<<<EW((size_t)BB * 5), 256, 0, stream>>>(f1, fc2w, fc2b, (float*)d_out);
}
